// TRIZRL_2508260901757
// MI455X (gfx1250) — compile-verified, matrix operand order checked by reading, not run on hardware
//
#include <hip/hip_runtime.h>
#include <hip/hip_bf16.h>

typedef __attribute__((ext_vector_type(16))) _Float16 v16h;
typedef __attribute__((ext_vector_type(8)))  float    v8f;

union Frag16 { v16h v; int4 q[2]; };

// ---------------------------------------------------------------------------
// Kernel 0: convert triz_w fp32 -> f16 (327KB result, stays L2-resident)
// ---------------------------------------------------------------------------
__global__ __launch_bounds__(256) void k_convert_w(const float* __restrict__ w,
                                                   _Float16* __restrict__ w16,
                                                   int n) {
    int i = blockIdx.x * blockDim.x + threadIdx.x;
    if (i < n) w16[i] = (_Float16)w[i];
}

// ---------------------------------------------------------------------------
// Kernel 1: encoder tanh(x@enc_w^T+b) -> z (stored f16 for WMMA),
//           policy logits -> softmax probs (output #2), categorical sample.
// ---------------------------------------------------------------------------
__global__ __launch_bounds__(256) void k_encode_policy(
    const float* __restrict__ x,
    const float* __restrict__ enc_w, const float* __restrict__ enc_b,
    const float* __restrict__ pol_w, const float* __restrict__ pol_b,
    _Float16* __restrict__ z16, float* __restrict__ probs,
    int* __restrict__ idx_out)
{
    const int b = blockIdx.x * blockDim.x + threadIdx.x;

    float xv[5];
#pragma unroll
    for (int i = 0; i < 5; ++i) xv[i] = x[(size_t)b * 5 + i];

    float z[64];
#pragma unroll
    for (int j = 0; j < 64; ++j) {
        float acc = enc_b[j];
#pragma unroll
        for (int i = 0; i < 5; ++i) acc = fmaf(xv[i], enc_w[j * 5 + i], acc);
        z[j] = tanhf(acc);
    }

    // pack z to f16 pairs, 32 dword stores
#pragma unroll
    for (int j = 0; j < 64; j += 2) {
        union { _Float16 h[2]; unsigned u; } p;
        p.h[0] = (_Float16)z[j];
        p.h[1] = (_Float16)z[j + 1];
        ((unsigned*)z16)[(size_t)b * 32 + (j >> 1)] = p.u;
    }

    // policy logits + softmax
    float l[40];
    float mx = -3.4e38f;
#pragma unroll
    for (int k = 0; k < 40; ++k) {
        float acc = pol_b[k];
#pragma unroll
        for (int j = 0; j < 64; ++j) acc = fmaf(z[j], pol_w[k * 64 + j], acc);
        l[k] = acc;
        mx = fmaxf(mx, acc);
    }
    float s = 0.0f;
#pragma unroll
    for (int k = 0; k < 40; ++k) { l[k] = __expf(l[k] - mx); s += l[k]; }
    const float inv = 1.0f / s;
#pragma unroll
    for (int k = 0; k < 40; ++k) probs[(size_t)b * 40 + k] = l[k] * inv;

    // deterministic categorical sample (PCG-style hash of row id), inverse CDF
    unsigned st = (unsigned)b * 747796405u + 2891336453u;
    unsigned wrd = ((st >> ((st >> 28) + 4u)) ^ st) * 277803737u;
    wrd = (wrd >> 22) ^ wrd;
    const float target = ((float)wrd * 2.3283064365386963e-10f) * s;
    float cum = 0.0f;
    int sel = -1;
#pragma unroll
    for (int k = 0; k < 40; ++k) {
        cum += l[k];
        if (sel < 0 && cum >= target) sel = k;
    }
    if (sel < 0) sel = 39;
    idx_out[b] = sel;
}

// ---------------------------------------------------------------------------
// Kernel 2: 40-expert batched GEMM + bias + relu via v_wmma_f32_16x16x32_f16.
// One wave = 16 batch rows x one expert's 64 outputs (4 col tiles, K=64).
// ---------------------------------------------------------------------------
__global__ __launch_bounds__(256) void k_experts_wmma(
    const _Float16* __restrict__ z16,   // [B,64] row-major f16
    const _Float16* __restrict__ w16,   // [40,64(o),64(d)] f16
    const float* __restrict__ triz_b,   // [40,64]
    float* __restrict__ h_out)          // [B,40,64]
{
    const int lane = threadIdx.x & 31;
    const int wave = threadIdx.x >> 5;
    const long wave_id = (long)blockIdx.x * 8 + wave;
    const int k    = (int)(wave_id % 40);
    const int tile = (int)(wave_id / 40);
    const int row0 = tile << 4;           // 16 rows per tile
    const int m    = lane & 15;
    const int hi   = lane >> 4;

    // A fragments: z rows, ISA 16-bit A layout (K pairs contiguous per half-wave)
    const _Float16* zrow = z16 + (size_t)(row0 + m) * 64;
    Frag16 a0, a1;
    a0.q[0] = *(const int4*)(zrow + hi * 8);
    a0.q[1] = *(const int4*)(zrow + hi * 8 + 16);
    a1.q[0] = *(const int4*)(zrow + 32 + hi * 8);
    a1.q[1] = *(const int4*)(zrow + 32 + hi * 8 + 16);

#pragma unroll
    for (int n0 = 0; n0 < 64; n0 += 16) {
        // B fragment: column n = lane&15 of W^T  == row (n0+m) of triz_w[k]
        const _Float16* wrow = w16 + ((size_t)k * 64 + (size_t)(n0 + m)) * 64;
        if (n0 < 48) __builtin_prefetch(wrow + 16 * 64, 0, 1);  // next col tile
        Frag16 b0, b1;
        b0.q[0] = *(const int4*)(wrow + hi * 8);
        b0.q[1] = *(const int4*)(wrow + hi * 8 + 16);
        b1.q[0] = *(const int4*)(wrow + 32 + hi * 8);
        b1.q[1] = *(const int4*)(wrow + 32 + hi * 8 + 16);

        v8f c = {};
        c = __builtin_amdgcn_wmma_f32_16x16x32_f16(false, a0.v, false, b0.v,
                                                   (short)0, c, false, false);
        c = __builtin_amdgcn_wmma_f32_16x16x32_f16(false, a1.v, false, b1.v,
                                                   (short)0, c, false, false);

        const float bias = triz_b[k * 64 + n0 + m];
#pragma unroll
        for (int r = 0; r < 8; ++r) {
            // C layout: element (M = r + 8*hi, N = m)
            const int M = r + 8 * hi;
            const float v = fmaxf(c[r] + bias, 0.0f);
            h_out[((size_t)(row0 + M) * 40 + k) * 64 + n0 + m] = v;
        }
    }
}

// ---------------------------------------------------------------------------
// Kernel 3: gather selected expert row + decoder Linear(64->5)
// ---------------------------------------------------------------------------
__global__ __launch_bounds__(256) void k_gather_decode(
    const float* __restrict__ h, const int* __restrict__ idx,
    const float* __restrict__ dec_w, const float* __restrict__ dec_b,
    float* __restrict__ out)
{
    const int b = blockIdx.x * blockDim.x + threadIdx.x;
    const int i = idx[b];
    const float* hr = h + ((size_t)b * 40 + (size_t)i) * 64;

    float acc[5];
#pragma unroll
    for (int o = 0; o < 5; ++o) acc[o] = dec_b[o];
#pragma unroll
    for (int j = 0; j < 64; ++j) {
        const float v = hr[j];
#pragma unroll
        for (int o = 0; o < 5; ++o) acc[o] = fmaf(v, dec_w[o * 64 + j], acc[o]);
    }
#pragma unroll
    for (int o = 0; o < 5; ++o) out[(size_t)b * 5 + o] = acc[o];
}

// ---------------------------------------------------------------------------
extern "C" void kernel_launch(void* const* d_in, const int* in_sizes, int n_in,
                              void* d_out, int out_size, void* d_ws, size_t ws_size,
                              hipStream_t stream) {
    const float* x      = (const float*)d_in[0];
    const float* enc_w  = (const float*)d_in[1];
    const float* enc_b  = (const float*)d_in[2];
    const float* triz_w = (const float*)d_in[3];
    const float* triz_b = (const float*)d_in[4];
    const float* pol_w  = (const float*)d_in[5];
    const float* pol_b  = (const float*)d_in[6];
    const float* dec_w  = (const float*)d_in[7];
    const float* dec_b  = (const float*)d_in[8];

    const int B = in_sizes[0] / 5;          // 131072

    float* out   = (float*)d_out;           // [B,5]
    float* probs = out + (size_t)B * 5;     // [B,40]
    float* h     = out + (size_t)B * 45;    // [B,40,64]

    char* ws = (char*)d_ws;
    _Float16* z16 = (_Float16*)ws;                                   // B*64 f16
    int*      idx = (int*)(ws + (size_t)B * 64 * 2);                 // B int
    _Float16* w16 = (_Float16*)(ws + (size_t)B * 64 * 2 + (size_t)B * 4); // 40*64*64 f16

    const int nW = 40 * 64 * 64;
    k_convert_w<<<(nW + 255) / 256, 256, 0, stream>>>(triz_w, w16, nW);

    k_encode_policy<<<B / 256, 256, 0, stream>>>(x, enc_w, enc_b, pol_w, pol_b,
                                                 z16, probs, idx);

    // (B/16) batch tiles * 40 experts waves, 8 waves (256 thr) per block
    const long waves  = (long)(B / 16) * 40;
    const int  blocks = (int)(waves / 8);
    k_experts_wmma<<<blocks, 256, 0, stream>>>(z16, w16, triz_b, h);

    k_gather_decode<<<B / 256, 256, 0, stream>>>(h, idx, dec_w, dec_b, out);
}
